// Block_30107720745811
// MI455X (gfx1250) — compile-verified, matrix operand order checked by reading, not run on hardware
//
#include <hip/hip_runtime.h>

// ---------------- problem constants (B=4, S=1024, H=1024, NH=16) ----------------
#define BATCH 4
#define SEQ   1024
#define HID   1024
#define NHEAD 16
#define DHEAD 64
#define FFN   4096
#define MTOK  (BATCH * SEQ)

typedef __attribute__((ext_vector_type(16))) __bf16 v16bf;
typedef __attribute__((ext_vector_type(8)))  float  v8f;
typedef __attribute__((ext_vector_type(4)))  unsigned int u32x4;
typedef __attribute__((ext_vector_type(2)))  unsigned int u32x2;

struct Frag32B { u32x4 lo, hi; };   // 32 bytes == one v16bf fragment

__device__ __forceinline__ __bf16 f2bf(float f) {
  unsigned u = __builtin_bit_cast(unsigned, f);
  u += 0x7FFFu + ((u >> 16) & 1u);          // round-to-nearest-even
  unsigned short h = (unsigned short)(u >> 16);
  return __builtin_bit_cast(__bf16, h);
}

__device__ __forceinline__ v16bf frag16(const __bf16* p_lo, const __bf16* p_hi) {
  Frag32B u;
  u.lo = *(const u32x4*)p_lo;
  u.hi = *(const u32x4*)p_hi;
  return __builtin_bit_cast(v16bf, u);
}

// CDNA5 async copy: global -> LDS, 16B per lane, tracked by ASYNCcnt.
__device__ __forceinline__ void async_copy_b128(unsigned lds_byte_addr, const void* gaddr) {
  asm volatile("global_load_async_to_lds_b128 %0, %1, off"
               :
               : "v"(lds_byte_addr), "v"((unsigned long long)(size_t)gaddr)
               : "memory");
}
__device__ __forceinline__ void wait_async0() {
  asm volatile("s_wait_asynccnt 0x0" ::: "memory");
}

// ---------------- fp32 -> bf16 convert ----------------
__global__ __launch_bounds__(256)
void cvt_f32_bf16(const float* __restrict__ in, __bf16* __restrict__ out, int n) {
  int i = blockIdx.x * 256 + threadIdx.x;
  if (i < n) out[i] = f2bf(in[i]);
}

// ---------------- tiled WMMA GEMM: C = A[MxK] @ B[KxN] + bias ----------------
// 8 waves (4x2), macrotile 128x64, each wave owns a 2x2 grid of 16x16 tiles.
// K-step 32, LDS double-buffered; A tile streamed with GLOBAL_LOAD_ASYNC_TO_LDS_B128,
// B tile VGPR-staged + transposed into LDS (2-byte scatter).
__global__ __launch_bounds__(256)
void gemm_bf16_wmma(const __bf16* __restrict__ A, const __bf16* __restrict__ B,
                    const float* __restrict__ bias,
                    float* __restrict__ Cf, __bf16* __restrict__ Cb,
                    int M, int N, int K, int relu)
{
  __shared__ __align__(16) __bf16 As[2 * 128 * 40];  // [buf][m][k], stride 40 halves
  __shared__ __align__(16) __bf16 Bs[2 * 64 * 40];   // [buf][n][k] (transposed), stride 40

  const int tid  = threadIdx.x;
  const int w    = tid >> 5;
  const int lane = tid & 31;
  const int lr   = lane & 15;
  const int hof  = (lane >> 4) ? 8 : 0;
  const int wr   = w >> 1;              // 0..3: 32-row band
  const int wc   = w & 1;               // 0..1: 32-col band
  const int m0   = blockIdx.y * 128;
  const int n0   = blockIdx.x * 64;

  // A-tile async issue: 128x32 halves = 512 b128 chunks, 2 per thread
  auto issueA = [&](int k0, int buf) {
    #pragma unroll
    for (int j = 0; j < 2; ++j) {
      int c = tid * 2 + j;
      int row = c >> 2, col = (c & 3) * 8;
      unsigned lds = (unsigned)(size_t)(As + buf * (128 * 40) + row * 40 + col);
      async_copy_b128(lds, A + (size_t)(m0 + row) * K + k0 + col);
    }
  };

  // B-tile: 32x64 halves, one b128 (8 halves) per thread, transposed on LDS store
  u32x4 breg;
  int bkr = (tid * 8) >> 6;        // k row 0..31
  int bnc = (tid * 8) & 63;        // n col, multiple of 8
  auto loadB = [&](int k0) {
    breg = *(const u32x4*)(B + (size_t)(k0 + bkr) * N + n0 + bnc);
  };
  auto storeB = [&](int buf) {
    const __bf16* tv = (const __bf16*)&breg;
    #pragma unroll
    for (int j = 0; j < 8; ++j) Bs[buf * (64 * 40) + (bnc + j) * 40 + bkr] = tv[j];
  };

  v8f acc00 = {}, acc01 = {}, acc10 = {}, acc11 = {};

  const int T = K >> 5;
  loadB(0);
  issueA(0, 0);
  storeB(0);
  wait_async0();
  __syncthreads();

  for (int t = 0; t < T; ++t) {
    const int cur = t & 1, nxt = cur ^ 1;
    const bool more = (t + 1) < T;
    if (more) { loadB((t + 1) * 32); issueA((t + 1) * 32, nxt); }

    // fragments (2 A rows-tiles, 2 B col-tiles), then 4 WMMA with full reuse
    const __bf16* ap0 = As + cur * (128 * 40) + (wr * 32 + lr) * 40;
    const __bf16* ap1 = ap0 + 16 * 40;
    const __bf16* bp0 = Bs + cur * (64 * 40) + (wc * 32 + lr) * 40;
    const __bf16* bp1 = bp0 + 16 * 40;
    v16bf a0 = frag16(ap0 + hof, ap0 + hof + 16);
    v16bf a1 = frag16(ap1 + hof, ap1 + hof + 16);
    v16bf b0 = frag16(bp0 + hof, bp0 + hof + 16);
    v16bf b1 = frag16(bp1 + hof, bp1 + hof + 16);
    acc00 = __builtin_amdgcn_wmma_f32_16x16x32_bf16(false, a0, false, b0, (short)0, acc00, false, false);
    acc01 = __builtin_amdgcn_wmma_f32_16x16x32_bf16(false, a0, false, b1, (short)0, acc01, false, false);
    acc10 = __builtin_amdgcn_wmma_f32_16x16x32_bf16(false, a1, false, b0, (short)0, acc10, false, false);
    acc11 = __builtin_amdgcn_wmma_f32_16x16x32_bf16(false, a1, false, b1, (short)0, acc11, false, false);

    if (more) {
      storeB(nxt);
      wait_async0();       // next tile's async A committed to LDS
      __syncthreads();     // next tile's B stores visible; safe to recycle buffers
    }
  }

  const int cn0 = n0 + wc * 32 + lr;
  const int cn1 = cn0 + 16;
  const float bv0 = bias ? bias[cn0] : 0.f;
  const float bv1 = bias ? bias[cn1] : 0.f;
  #pragma unroll
  for (int i = 0; i < 8; ++i) {
    int r0 = m0 + wr * 32 + hof + i;
    int r1 = r0 + 16;
    float v00 = acc00[i] + bv0, v01 = acc01[i] + bv1;
    float v10 = acc10[i] + bv0, v11 = acc11[i] + bv1;
    if (relu) {
      v00 = fmaxf(v00, 0.f); v01 = fmaxf(v01, 0.f);
      v10 = fmaxf(v10, 0.f); v11 = fmaxf(v11, 0.f);
    }
    if (Cf) {
      Cf[(size_t)r0 * N + cn0] = v00; Cf[(size_t)r0 * N + cn1] = v01;
      Cf[(size_t)r1 * N + cn0] = v10; Cf[(size_t)r1 * N + cn1] = v11;
    }
    if (Cb) {
      Cb[(size_t)r0 * N + cn0] = f2bf(v00); Cb[(size_t)r0 * N + cn1] = f2bf(v01);
      Cb[(size_t)r1 * N + cn0] = f2bf(v10); Cb[(size_t)r1 * N + cn1] = f2bf(v11);
    }
  }
}

// ---------------- flash attention (bf16 QKV in, bf16 out) ----------------
// 4 waves per block; each wave owns one 16-row q-subtile of a 64-row q-tile.
// softmax((q.k/scale + add)/scale) == softmax(q.k/scale^2 + add/scale)
__global__ __launch_bounds__(128)
void flash_attn_bf16(const __bf16* __restrict__ Q, const __bf16* __restrict__ Kd,
                     const __bf16* __restrict__ V, const int* __restrict__ mask,
                     __bf16* __restrict__ O)
{
  __shared__ __align__(16) __bf16 Ks[32 * 72];      // [kv][d], stride 72 (144B rows)
  __shared__ __align__(16) __bf16 Vs[64 * 40];      // transposed: [n][kv]
  __shared__ __align__(16) __bf16 Ps[4 * 16 * 40];  // per-wave P tile (C->A relayout)

  const int tid  = threadIdx.x;
  const int w    = tid >> 5;
  const int lane = tid & 31;
  const int lr   = lane & 15;
  const int hof  = (lane >> 4) ? 8 : 0;
  const int q0   = blockIdx.x * 64;
  const int h    = blockIdx.y;
  const int bb   = blockIdx.z;
  const size_t base = ((size_t)bb * SEQ) * HID + (size_t)h * DHEAD;

  v16bf qf0, qf1;
  {
    const __bf16* qp = Q + base + (size_t)(q0 + w * 16 + lr) * HID;
    qf0 = frag16(qp + hof,      qp + hof + 16);
    qf1 = frag16(qp + 32 + hof, qp + 32 + hof + 16);
  }

  const v8f vzero = {};
  v8f o0 = vzero, o1 = vzero, o2 = vzero, o3 = vzero;
  float m_i[8], l_i[8];
  #pragma unroll
  for (int i = 0; i < 8; ++i) { m_i[i] = -3.0e38f; l_i[i] = 0.f; }

  const float inv_s2 = 1.0f / 64.0f;     // 1/scale^2, scale = sqrt(64) = 8
  const float mneg   = -1.25e9f;         // (-1e10)/scale

  for (int kb0 = 0; kb0 < SEQ; kb0 += 32) {
    #pragma unroll
    for (int j = 0; j < 2; ++j) {
      int c = tid * 2 + j;
      int row = c >> 3, col = (c & 7) * 8;
      *(u32x4*)(Ks + row * 72 + col) =
        *(const u32x4*)(Kd + base + (size_t)(kb0 + row) * HID + col);
    }
    {
      int kv = tid >> 2, nseg = (tid & 3) * 16;
      Frag32B u;
      u.lo = *(const u32x4*)(V + base + (size_t)(kb0 + kv) * HID + nseg);
      u.hi = *(const u32x4*)(V + base + (size_t)(kb0 + kv) * HID + nseg + 8);
      const __bf16* tv = (const __bf16*)&u;
      #pragma unroll
      for (int e = 0; e < 16; ++e) Vs[(nseg + e) * 40 + kv] = tv[e];
    }
    __syncthreads();

    v8f s0 = vzero, s1 = vzero;
    {
      const __bf16* k0p = Ks + lr * 72;
      const __bf16* k1p = Ks + (16 + lr) * 72;
      v16bf b00 = frag16(k0p + hof,      k0p + hof + 16);
      v16bf b01 = frag16(k0p + 32 + hof, k0p + 32 + hof + 16);
      v16bf b10 = frag16(k1p + hof,      k1p + hof + 16);
      v16bf b11 = frag16(k1p + 32 + hof, k1p + 32 + hof + 16);
      s0 = __builtin_amdgcn_wmma_f32_16x16x32_bf16(false, qf0, false, b00, (short)0, s0, false, false);
      s0 = __builtin_amdgcn_wmma_f32_16x16x32_bf16(false, qf1, false, b01, (short)0, s0, false, false);
      s1 = __builtin_amdgcn_wmma_f32_16x16x32_bf16(false, qf0, false, b10, (short)0, s1, false, false);
      s1 = __builtin_amdgcn_wmma_f32_16x16x32_bf16(false, qf1, false, b11, (short)0, s1, false, false);
    }

    const float a0 = mask[bb * SEQ + kb0 + lr]      ? 0.f : mneg;
    const float a1 = mask[bb * SEQ + kb0 + 16 + lr] ? 0.f : mneg;

    #pragma unroll
    for (int i = 0; i < 8; ++i) {
      float sv0 = s0[i] * inv_s2 + a0;
      float sv1 = s1[i] * inv_s2 + a1;
      float mx = fmaxf(sv0, sv1);
      #pragma unroll
      for (int d = 1; d < 16; d <<= 1) mx = fmaxf(mx, __shfl_xor(mx, d, 32));
      float mnew = fmaxf(m_i[i], mx);
      float corr = __expf(m_i[i] - mnew);
      float e0 = __expf(sv0 - mnew);
      float e1 = __expf(sv1 - mnew);
      float rs = e0 + e1;
      #pragma unroll
      for (int d = 1; d < 16; d <<= 1) rs += __shfl_xor(rs, d, 32);
      l_i[i] = l_i[i] * corr + rs;
      m_i[i] = mnew;
      o0[i] *= corr; o1[i] *= corr; o2[i] *= corr; o3[i] *= corr;
      Ps[(w * 16 + hof + i) * 40 + lr]      = f2bf(e0);
      Ps[(w * 16 + hof + i) * 40 + 16 + lr] = f2bf(e1);
    }
    __syncthreads();

    {
      const __bf16* pp = Ps + (w * 16 + lr) * 40;
      v16bf af = frag16(pp + hof, pp + hof + 16);
      const __bf16* v0p = Vs + lr * 40;
      const __bf16* v1p = Vs + (16 + lr) * 40;
      const __bf16* v2p = Vs + (32 + lr) * 40;
      const __bf16* v3p = Vs + (48 + lr) * 40;
      o0 = __builtin_amdgcn_wmma_f32_16x16x32_bf16(false, af, false, frag16(v0p + hof, v0p + hof + 16), (short)0, o0, false, false);
      o1 = __builtin_amdgcn_wmma_f32_16x16x32_bf16(false, af, false, frag16(v1p + hof, v1p + hof + 16), (short)0, o1, false, false);
      o2 = __builtin_amdgcn_wmma_f32_16x16x32_bf16(false, af, false, frag16(v2p + hof, v2p + hof + 16), (short)0, o2, false, false);
      o3 = __builtin_amdgcn_wmma_f32_16x16x32_bf16(false, af, false, frag16(v3p + hof, v3p + hof + 16), (short)0, o3, false, false);
    }
    __syncthreads();
  }

  #pragma unroll
  for (int i = 0; i < 8; ++i) {
    float invl = 1.0f / l_i[i];
    size_t rowb = base + (size_t)(q0 + w * 16 + hof + i) * HID;
    O[rowb +  0 + lr] = f2bf(o0[i] * invl);
    O[rowb + 16 + lr] = f2bf(o1[i] * invl);
    O[rowb + 32 + lr] = f2bf(o2[i] * invl);
    O[rowb + 48 + lr] = f2bf(o3[i] * invl);
  }
}

// ---------------- fused residual-add + LayerNorm (f32 out + optional bf16 out) ----------------
__global__ __launch_bounds__(256)
void add_layernorm(const float* __restrict__ x, const float* __restrict__ res,
                   const float* __restrict__ gw, const float* __restrict__ bw,
                   float* __restrict__ y, __bf16* __restrict__ yb)
{
  const size_t rb = (size_t)blockIdx.x * HID;
  float vals[4], s = 0.f, ss = 0.f;
  #pragma unroll
  for (int j = 0; j < 4; ++j) {
    int col = threadIdx.x + j * 256;
    float v = x[rb + col] + res[rb + col];
    vals[j] = v; s += v; ss += v * v;
  }
  #pragma unroll
  for (int d = 1; d < 32; d <<= 1) { s += __shfl_xor(s, d, 32); ss += __shfl_xor(ss, d, 32); }
  __shared__ float shs[8], shss[8];
  if ((threadIdx.x & 31) == 0) { shs[threadIdx.x >> 5] = s; shss[threadIdx.x >> 5] = ss; }
  __syncthreads();
  s = 0.f; ss = 0.f;
  #pragma unroll
  for (int i = 0; i < 8; ++i) { s += shs[i]; ss += shss[i]; }
  const float mean = s * (1.0f / HID);
  const float var  = ss * (1.0f / HID) - mean * mean;
  const float inv  = rsqrtf(var + 1e-5f);
  #pragma unroll
  for (int j = 0; j < 4; ++j) {
    int col = threadIdx.x + j * 256;
    float o = gw[col] * (vals[j] - mean) * inv + bw[col];
    y[rb + col] = o;
    if (yb) yb[rb + col] = f2bf(o);
  }
}

// ---------------- host orchestration ----------------
extern "C" void kernel_launch(void* const* d_in, const int* in_sizes, int n_in,
                              void* d_out, int out_size, void* d_ws, size_t ws_size,
                              hipStream_t stream) {
  (void)in_sizes; (void)n_in; (void)out_size; (void)ws_size;
  const int M = MTOK, H = HID, F = FFN;

  const float* hs    = (const float*)d_in[0];
  const int*   smask = (const int*)  d_in[1];
  const float* chs   = (const float*)d_in[2];
  const int*   cmask = (const int*)  d_in[3];
  const float* sWq = (const float*)d_in[5],  *sbq = (const float*)d_in[6];
  const float* sWk = (const float*)d_in[7],  *sbk = (const float*)d_in[8];
  const float* sWv = (const float*)d_in[9],  *sbv = (const float*)d_in[10];
  const float* sWo = (const float*)d_in[11], *sbo = (const float*)d_in[12];
  const float* cWq = (const float*)d_in[13], *cbq = (const float*)d_in[14];
  const float* cWk = (const float*)d_in[15], *cbk = (const float*)d_in[16];
  const float* cWv = (const float*)d_in[17], *cbv = (const float*)d_in[18];
  const float* cWo = (const float*)d_in[19], *cbo = (const float*)d_in[20];
  const float* gln = (const float*)d_in[21], *bln = (const float*)d_in[22];
  const float* W1  = (const float*)d_in[23], *b1  = (const float*)d_in[24];
  const float* W2  = (const float*)d_in[25], *b2  = (const float*)d_in[26];

  char* wsp = (char*)d_ws;
  size_t off = 0;
  auto alloc = [&](size_t bytes) -> void* {
    void* p = wsp + off;
    off = (off + bytes + 255) & ~(size_t)255;
    return p;
  };

  __bf16* wqb  = (__bf16*)alloc((size_t)H * H * 2);
  __bf16* wkb  = (__bf16*)alloc((size_t)H * H * 2);
  __bf16* wvb  = (__bf16*)alloc((size_t)H * H * 2);
  __bf16* wob  = (__bf16*)alloc((size_t)H * H * 2);
  __bf16* cwqb = (__bf16*)alloc((size_t)H * H * 2);
  __bf16* cwkb = (__bf16*)alloc((size_t)H * H * 2);
  __bf16* cwvb = (__bf16*)alloc((size_t)H * H * 2);
  __bf16* cwob = (__bf16*)alloc((size_t)H * H * 2);
  __bf16* w1b  = (__bf16*)alloc((size_t)H * F * 2);
  __bf16* w2b  = (__bf16*)alloc((size_t)F * H * 2);
  __bf16* xbf  = (__bf16*)alloc((size_t)M * H * 2);
  __bf16* xcbf = (__bf16*)alloc((size_t)M * H * 2);
  __bf16* qb   = (__bf16*)alloc((size_t)M * H * 2);
  __bf16* kb   = (__bf16*)alloc((size_t)M * H * 2);
  __bf16* vb   = (__bf16*)alloc((size_t)M * H * 2);
  __bf16* ab   = (__bf16*)alloc((size_t)M * H * 2);
  float*  t0   = (float*) alloc((size_t)M * H * 4);
  float*  ln1  = (float*) alloc((size_t)M * H * 4);
  __bf16* ln1b = (__bf16*)alloc((size_t)M * H * 2);
  float*  ln2  = (float*) alloc((size_t)M * H * 4);
  __bf16* ln2b = (__bf16*)alloc((size_t)M * H * 2);
  __bf16* ff1b = (__bf16*)alloc((size_t)M * F * 2);

  auto cvt = [&](const float* src, __bf16* dst, int n) {
    cvt_f32_bf16<<<(n + 255) / 256, 256, 0, stream>>>(src, dst, n);
  };
  cvt(hs,  xbf,  M * H);
  cvt(chs, xcbf, M * H);
  cvt(sWq, wqb,  H * H); cvt(sWk, wkb, H * H); cvt(sWv, wvb, H * H); cvt(sWo, wob, H * H);
  cvt(cWq, cwqb, H * H); cvt(cWk, cwkb, H * H); cvt(cWv, cwvb, H * H); cvt(cWo, cwob, H * H);
  cvt(W1,  w1b,  H * F); cvt(W2,  w2b,  F * H);

  const dim3 gH(H / 64, M / 128);
  const dim3 gF(F / 64, M / 128);
  const dim3 gA(SEQ / 64, NHEAD, BATCH);

  // ---- self attention ----
  gemm_bf16_wmma<<<gH, 256, 0, stream>>>(xbf, wqb, sbq, nullptr, qb, M, H, H, 0);
  gemm_bf16_wmma<<<gH, 256, 0, stream>>>(xbf, wkb, sbk, nullptr, kb, M, H, H, 0);
  gemm_bf16_wmma<<<gH, 256, 0, stream>>>(xbf, wvb, sbv, nullptr, vb, M, H, H, 0);
  flash_attn_bf16<<<gA, 128, 0, stream>>>(qb, kb, vb, smask, ab);
  gemm_bf16_wmma<<<gH, 256, 0, stream>>>(ab, wob, sbo, t0, nullptr, M, H, H, 0);
  add_layernorm<<<M, 256, 0, stream>>>(t0, hs, gln, bln, ln1, ln1b);

  // ---- cross attention (residual from ORIGINAL hidden_states, per reference) ----
  gemm_bf16_wmma<<<gH, 256, 0, stream>>>(ln1b, cwqb, cbq, nullptr, qb, M, H, H, 0);
  gemm_bf16_wmma<<<gH, 256, 0, stream>>>(xcbf, cwkb, cbk, nullptr, kb, M, H, H, 0);
  gemm_bf16_wmma<<<gH, 256, 0, stream>>>(xcbf, cwvb, cbv, nullptr, vb, M, H, H, 0);
  flash_attn_bf16<<<gA, 128, 0, stream>>>(qb, kb, vb, cmask, ab);
  gemm_bf16_wmma<<<gH, 256, 0, stream>>>(ab, cwob, cbo, t0, nullptr, M, H, H, 0);
  add_layernorm<<<M, 256, 0, stream>>>(t0, hs, gln, bln, ln2, ln2b);

  // ---- FFN ----
  gemm_bf16_wmma<<<gF, 256, 0, stream>>>(ln2b, w1b, b1, nullptr, ff1b, M, F, H, 1);
  gemm_bf16_wmma<<<gH, 256, 0, stream>>>(ff1b, w2b, b2, t0, nullptr, M, H, F, 0);
  add_layernorm<<<M, 256, 0, stream>>>(t0, ln2, gln, bln, (float*)d_out, nullptr);
}
